// CascadedAttention_21947282883218
// MI455X (gfx1250) — matrix operand order repaired and checked by reading, not compiled, not run on hardware
//
#include <hip/hip_runtime.h>
#include <hip/hip_bf16.h>
#include <stdint.h>

// ---------------- problem constants ----------------
#define BB 64
#define TT 512
#define DD 768
#define HH 768
#define VV 28
#define KTOT 2304   // e | state | ctx  along K
#define NZR 1536    // z | r gate columns

typedef __attribute__((ext_vector_type(16))) __bf16 v16bf;
typedef __attribute__((ext_vector_type(8)))  float  v8f;

union Frag16 { v16bf v; uint4 q[2]; };

__device__ __forceinline__ unsigned short f32_to_bf16(float f) {
    unsigned int u = __float_as_uint(f);
    unsigned int r = u + 0x7FFFu + ((u >> 16) & 1u);   // round-to-nearest-even
    return (unsigned short)(r >> 16);
}

__device__ __forceinline__ float fast_tanh(float x) {
    if (x >  8.f) return  1.f;
    if (x < -8.f) return -1.f;
    float t = __expf(2.f * x);
    return (t - 1.f) / (t + 1.f);
}

__device__ __forceinline__ float fast_sigmoid(float x) {
    return 1.f / (1.f + __expf(-x));
}

// ---------------- weight repack: fp32 [K][N] blocks -> bf16 transposed [N][KTOT] ----------------
__global__ __launch_bounds__(256) void k_pack_zr(
    const float* __restrict__ Wz, const float* __restrict__ Uz, const float* __restrict__ Cz,
    const float* __restrict__ Wr, const float* __restrict__ Ur, const float* __restrict__ Cr,
    unsigned short* __restrict__ wzrT)
{
    int idx = blockIdx.x * 256 + threadIdx.x;
    if (idx >= NZR * KTOT) return;
    int n = idx / KTOT, k = idx % KTOT;
    bool zg = n < HH;
    int nn = zg ? n : n - HH;
    int blk = k / HH, kk = k % HH;
    const float* W = (blk == 0) ? (zg ? Wz : Wr) : (blk == 1) ? (zg ? Uz : Ur) : (zg ? Cz : Cr);
    wzrT[(size_t)n * KTOT + k] = f32_to_bf16(W[(size_t)kk * HH + nn]);
}

__global__ __launch_bounds__(256) void k_pack_h(
    const float* __restrict__ Wh, const float* __restrict__ Uh, const float* __restrict__ Ch,
    unsigned short* __restrict__ whT)
{
    int idx = blockIdx.x * 256 + threadIdx.x;
    if (idx >= HH * KTOT) return;
    int n = idx / KTOT, k = idx % KTOT;
    int blk = k / HH, kk = k % HH;
    const float* W = (blk == 0) ? Wh : (blk == 1) ? Uh : Ch;
    whT[(size_t)n * KTOT + k] = f32_to_bf16(W[(size_t)kk * HH + n]);
}

// ---------------- init ----------------
__global__ __launch_bounds__(256) void k_init(float* __restrict__ state,
                                              unsigned short* __restrict__ s16,
                                              int* __restrict__ pred_idx)
{
    int idx = blockIdx.x * 256 + threadIdx.x;
    if (idx < BB * HH) { state[idx] = 0.f; s16[idx] = 0; }
    if (idx < BB) pred_idx[idx] = 0;
}

// ---------------- attention phase 1: scores for one (b, 64-timestep slice) ----------------
__global__ __launch_bounds__(256) void k_scores(
    const float* __restrict__ x,        // [B,T,D]
    const float* __restrict__ scale,    // [D]
    const float* __restrict__ state,    // [B,H]
    float* __restrict__ scores)         // [B,T]
{
    __shared__ float s_state[DD];
    __shared__ float s_scale[DD];
    const int b  = blockIdx.x;
    const int t0 = blockIdx.y * 64;     // 8 slices of 64 timesteps
    const int tid = threadIdx.x;
    for (int d = tid; d < DD; d += 256) { s_state[d] = state[b * DD + d]; s_scale[d] = scale[d]; }
    __syncthreads();

    const int wave = tid >> 5, lane = tid & 31;
    const float* xb = x + (size_t)b * TT * DD;
    for (int t = t0 + wave; t < t0 + 64; t += 8) {
        const float* xt = xb + (size_t)t * DD;
        __builtin_prefetch(xt + (size_t)8 * DD + lane * 4, 0, 0);  // next row this wave touches
        float acc = 0.f;
        for (int d = lane; d < DD; d += 32)
            acc += s_scale[d] * fast_tanh(s_state[d] + xt[d]);
        for (int m = 16; m; m >>= 1) acc += __shfl_xor(acc, m, 32);
        if (lane == 0) scores[b * TT + t] = acc;
    }
}

// ---------------- attention phase 2: softmax over T + context (bf16) ----------------
__global__ __launch_bounds__(256) void k_softmax_ctx(
    const float* __restrict__ x,        // [B,T,D]
    const float* __restrict__ scores,   // [B,T]
    unsigned short* __restrict__ c16)   // [B,D]
{
    __shared__ float s_sc[TT];
    __shared__ float s_red[256];
    const int b = blockIdx.x;
    const int tid = threadIdx.x;

    float a0 = scores[b * TT + tid], a1 = scores[b * TT + tid + 256];
    s_red[tid] = fmaxf(a0, a1);
    __syncthreads();
    for (int s = 128; s; s >>= 1) { if (tid < s) s_red[tid] = fmaxf(s_red[tid], s_red[tid + s]); __syncthreads(); }
    float mx = s_red[0];
    __syncthreads();
    float e0 = __expf(a0 - mx), e1 = __expf(a1 - mx);
    s_red[tid] = e0 + e1;
    __syncthreads();
    for (int s = 128; s; s >>= 1) { if (tid < s) s_red[tid] += s_red[tid + s]; __syncthreads(); }
    float inv = 1.f / s_red[0];
    s_sc[tid] = e0 * inv; s_sc[tid + 256] = e1 * inv;
    __syncthreads();

    const float* xb = x + (size_t)b * TT * DD;
    for (int d = tid; d < DD; d += 256) {
        float acc = 0.f;
        for (int t = 0; t < TT; ++t) acc += s_sc[t] * xb[(size_t)t * DD + d];
        c16[b * DD + d] = f32_to_bf16(acc);
    }
}

// ---------------- embedding gather (uses previous step's argmax) ----------------
__global__ __launch_bounds__(256) void k_gather_e(
    const float* __restrict__ emb,      // [V+1, H]
    const int* __restrict__ pred_idx,   // [B]
    unsigned short* __restrict__ e16,   // [B,H]
    int t)
{
    int idx = blockIdx.x * 256 + threadIdx.x;
    if (idx >= BB * HH) return;
    int b = idx / HH, h = idx % HH;
    int row = (t == 0) ? 0 : (pred_idx[b] + 1);
    e16[idx] = f32_to_bf16(emb[(size_t)row * HH + h]);
}

// ---------------- pred = state @ Wd + bd ; argmax ; write output row ----------------
__global__ __launch_bounds__(896) void k_pred(
    const float* __restrict__ state, const float* __restrict__ Wd, const float* __restrict__ bd,
    float* __restrict__ out, int* __restrict__ pred_idx, int t)
{
    __shared__ float s_pred[VV];
    const int b = blockIdx.x;
    const int wave = threadIdx.x >> 5, lane = threadIdx.x & 31;   // 28 waves, one per vocab col
    const float* sb = state + b * HH;
    float acc = 0.f;
    for (int k = lane; k < HH; k += 32) acc += sb[k] * Wd[(size_t)k * VV + wave];
    for (int m = 16; m; m >>= 1) acc += __shfl_xor(acc, m, 32);
    if (lane == 0) s_pred[wave] = acc + bd[wave];
    __syncthreads();
    if (threadIdx.x < VV) out[((size_t)b * TT + t) * VV + threadIdx.x] = s_pred[threadIdx.x];
    if (threadIdx.x == 0) {
        float best = s_pred[0]; int bi = 0;
        for (int v = 1; v < VV; ++v) if (s_pred[v] > best) { best = s_pred[v]; bi = v; }
        pred_idx[b] = bi;
    }
}

// ---------------- fragment loaders ----------------
__device__ __forceinline__ void load_a(
    const unsigned short* __restrict__ A0, const unsigned short* __restrict__ A1,
    const unsigned short* __restrict__ A2, int k0, int rowA, int kA, Frag16& a)
{
    const unsigned short* Ap = (k0 < HH) ? A0 : (k0 < 2 * HH) ? A1 : A2;
    const int kk = k0 - ((k0 < HH) ? 0 : (k0 < 2 * HH) ? HH : 2 * HH);
    const unsigned short* ap = Ap + (size_t)rowA * HH + kk + kA;
    a.q[0] = *reinterpret_cast<const uint4*>(ap);
    a.q[1] = *reinterpret_cast<const uint4*>(ap + 16);
}

__device__ __forceinline__ void load_b(const unsigned short* __restrict__ Bp, int k0, int kB, Frag16& b)
{
    const unsigned short* bp = Bp + k0 + kB;
    b.q[0] = *reinterpret_cast<const uint4*>(bp);
    b.q[1] = *reinterpret_cast<const uint4*>(bp + 16);
}

// ---------------- bf16 WMMA GEMM, software-pipelined K loop ----------------
// D[64,Ntot] = [A0|A1|A2][64,2304] * Bt^T ; Bt: [Ntot][2304] bf16.
// Each wave: one 16-row M tile x four 16-col N tiles; next-K fragments prefetched
// into a second register set so WMMAs overlap the loads.
__global__ __launch_bounds__(128) void k_gemm_wmma(
    const unsigned short* __restrict__ A0,
    const unsigned short* __restrict__ A1,
    const unsigned short* __restrict__ A2,
    const unsigned short* __restrict__ Bt,
    float* __restrict__ D, int Ntot)
{
    const int wave = threadIdx.x >> 5;
    const int lane = threadIdx.x & 31;
    const int task = blockIdx.x * 4 + wave;
    const int mtile = task & 3;
    const int nbase = (task >> 2) * 64;
    if (nbase >= Ntot) return;               // wave-uniform

    const int m0 = mtile * 16;
    const int nlane = lane & 15;
    const int half = lane >> 4;
    const int rowA = m0 + nlane;
    const int kA = half * 8;                 // A: K groups {0..7,16..23} / {8..15,24..31}
    const int kB = half * 8;                 // B: K groups {0..7,16..23} / {8..15,24..31}

    const unsigned short* Bp0 = Bt + (size_t)(nbase +  0 + nlane) * KTOT;
    const unsigned short* Bp1 = Bt + (size_t)(nbase + 16 + nlane) * KTOT;
    const unsigned short* Bp2 = Bt + (size_t)(nbase + 32 + nlane) * KTOT;
    const unsigned short* Bp3 = Bt + (size_t)(nbase + 48 + nlane) * KTOT;

    v8f acc0 = {}, acc1 = {}, acc2 = {}, acc3 = {};

    Frag16 a, b0, b1, b2, b3;                // current
    Frag16 an, bn0, bn1, bn2, bn3;           // next (prefetch)

    load_a(A0, A1, A2, 0, rowA, kA, a);
    load_b(Bp0, 0, kB, b0);
    load_b(Bp1, 0, kB, b1);
    load_b(Bp2, 0, kB, b2);
    load_b(Bp3, 0, kB, b3);

    #pragma unroll 2
    for (int k0 = 0; k0 < KTOT; k0 += 32) {
        const int k1 = k0 + 32;
        if (k1 < KTOT) {
            load_a(A0, A1, A2, k1, rowA, kA, an);
            load_b(Bp0, k1, kB, bn0);
            load_b(Bp1, k1, kB, bn1);
            load_b(Bp2, k1, kB, bn2);
            load_b(Bp3, k1, kB, bn3);
        }
        acc0 = __builtin_amdgcn_wmma_f32_16x16x32_bf16(false, a.v, false, b0.v, (short)0, acc0, false, false);
        acc1 = __builtin_amdgcn_wmma_f32_16x16x32_bf16(false, a.v, false, b1.v, (short)0, acc1, false, false);
        acc2 = __builtin_amdgcn_wmma_f32_16x16x32_bf16(false, a.v, false, b2.v, (short)0, acc2, false, false);
        acc3 = __builtin_amdgcn_wmma_f32_16x16x32_bf16(false, a.v, false, b3.v, (short)0, acc3, false, false);
        a = an; b0 = bn0; b1 = bn1; b2 = bn2; b3 = bn3;
    }

    const int mrow = m0 + half * 8;          // C/D: VGPR i -> row m0 + i (+8 for upper half)
    for (int i = 0; i < 8; ++i) {
        float* dr = D + (size_t)(mrow + i) * Ntot + nbase + nlane;
        dr[0]  = acc0[i];
        dr[16] = acc1[i];
        dr[32] = acc2[i];
        dr[48] = acc3[i];
    }
}

// ---------------- z/r activation + state*r (bf16) ----------------
__global__ __launch_bounds__(256) void k_act_zr(
    const float* __restrict__ zr_pre,   // [B,1536]
    const float* __restrict__ bz, const float* __restrict__ br,
    const float* __restrict__ state,
    float* __restrict__ zbuf, unsigned short* __restrict__ sr16)
{
    int idx = blockIdx.x * 256 + threadIdx.x;
    if (idx >= BB * HH) return;
    int b = idx / HH, h = idx % HH;
    float z = fast_sigmoid(zr_pre[(size_t)b * NZR + h] + bz[h]);
    float r = fast_sigmoid(zr_pre[(size_t)b * NZR + HH + h] + br[h]);
    zbuf[idx] = z;
    sr16[idx] = f32_to_bf16(state[idx] * r);
}

// ---------------- candidate tanh + GRU blend; update state + bf16 copy ----------------
__global__ __launch_bounds__(256) void k_update(
    const float* __restrict__ h_pre,    // [B,768]
    const float* __restrict__ zbuf,
    float* __restrict__ state, unsigned short* __restrict__ s16)
{
    int idx = blockIdx.x * 256 + threadIdx.x;
    if (idx >= BB * HH) return;
    float hh = fast_tanh(h_pre[idx]);
    float z  = zbuf[idx];
    float hn = (1.f - z) * hh + z * state[idx];
    state[idx] = hn;
    s16[idx] = f32_to_bf16(hn);
}

// ---------------- host launcher ----------------
extern "C" void kernel_launch(void* const* d_in, const int* in_sizes, int n_in,
                              void* d_out, int out_size, void* d_ws, size_t ws_size,
                              hipStream_t stream) {
    const float* inputs = (const float*)d_in[0];
    const float* emb    = (const float*)d_in[1];
    const float* Wz = (const float*)d_in[2];
    const float* Uz = (const float*)d_in[3];
    const float* Cz = (const float*)d_in[4];
    const float* Wr = (const float*)d_in[5];
    const float* Ur = (const float*)d_in[6];
    const float* Cr = (const float*)d_in[7];
    const float* Wh = (const float*)d_in[8];
    const float* Uh = (const float*)d_in[9];
    const float* Ch = (const float*)d_in[10];
    const float* bz = (const float*)d_in[11];
    const float* br = (const float*)d_in[12];
    const float* scale = (const float*)d_in[13];
    const float* Wd = (const float*)d_in[14];
    const float* bd = (const float*)d_in[15];
    float* out = (float*)d_out;

    // ---- workspace carve (all offsets 256B aligned) ----
    char* ws = (char*)d_ws;
    size_t off = 0;
    auto carve = [&](size_t bytes) { void* p = ws + off; off += (bytes + 255) & ~(size_t)255; return p; };
    unsigned short* wzrT = (unsigned short*)carve((size_t)NZR * KTOT * 2);  // 7.08 MB
    unsigned short* whT  = (unsigned short*)carve((size_t)HH  * KTOT * 2);  // 3.54 MB
    unsigned short* e16  = (unsigned short*)carve((size_t)BB * HH * 2);
    unsigned short* s16  = (unsigned short*)carve((size_t)BB * HH * 2);
    unsigned short* c16  = (unsigned short*)carve((size_t)BB * HH * 2);
    unsigned short* sr16 = (unsigned short*)carve((size_t)BB * HH * 2);
    float* state  = (float*)carve((size_t)BB * HH * 4);
    float* zbuf   = (float*)carve((size_t)BB * HH * 4);
    float* zr_pre = (float*)carve((size_t)BB * NZR * 4);
    float* h_pre  = (float*)carve((size_t)BB * HH * 4);
    float* sc_buf = (float*)carve((size_t)BB * TT * 4);                     // attention scores
    int*   pidx   = (int*)carve((size_t)BB * 4);
    (void)ws_size; (void)in_sizes; (void)n_in; (void)out_size;

    // ---- one-time (per launch) weight repack to bf16 transposed ----
    k_pack_zr<<<(NZR * KTOT + 255) / 256, 256, 0, stream>>>(Wz, Uz, Cz, Wr, Ur, Cr, wzrT);
    k_pack_h <<<(HH  * KTOT + 255) / 256, 256, 0, stream>>>(Wh, Uh, Ch, whT);
    k_init   <<<(BB * HH + 255) / 256, 256, 0, stream>>>(state, s16, pidx);

    const int elemBlocks = (BB * HH + 255) / 256;   // 192
    const int zrBlocks = (4 * (NZR / 64)) / 4;      // 96 wave-tasks / 4 waves = 24
    const int hBlocks  = (4 * (HH  / 64)) / 4;      // 48 wave-tasks / 4 waves = 12
    const dim3 scGrid(BB, 8);                       // 512 blocks for the tanh-heavy phase

    for (int t = 0; t < TT; ++t) {
        k_scores     <<<scGrid, 256, 0, stream>>>(inputs, scale, state, sc_buf);
        k_softmax_ctx<<<BB, 256, 0, stream>>>(inputs, sc_buf, c16);
        k_gather_e   <<<elemBlocks, 256, 0, stream>>>(emb, pidx, e16, t);   // reads pidx from t-1
        k_pred       <<<BB, 896, 0, stream>>>(state, Wd, bd, out, pidx, t); // overwrites pidx
        k_gemm_wmma  <<<zrBlocks, 128, 0, stream>>>(e16, s16,  c16, wzrT, zr_pre, NZR);
        k_act_zr     <<<elemBlocks, 256, 0, stream>>>(zr_pre, bz, br, state, zbuf, sr16);
        k_gemm_wmma  <<<hBlocks, 128, 0, stream>>>(e16, sr16, c16, whT,  h_pre,  HH);
        k_update     <<<elemBlocks, 256, 0, stream>>>(h_pre, zbuf, state, s16);
    }
}
